// SimpleRoPEAttention_27865747816948
// MI455X (gfx1250) — matrix operand order repaired and checked by reading, not compiled, not run on hardware
//
#include <hip/hip_runtime.h>
#include <hip/hip_bf16.h>
#include <cstdint>

// SimpleRoPEAttention on MI455X (gfx1250): QKV GEMM + RoPE + flash attention.
// Matmuls on v_wmma_f32_16x16x32_f16 (wave32); shared operand tiles staged in
// LDS with double-buffered global_load_async_to_lds_b128 (ASYNCcnt pipeline).
//
// B=2, S=2048, D=1024, H=16, HD=64.

typedef _Float16 half_t;
typedef __attribute__((ext_vector_type(16))) _Float16 v16h;
typedef __attribute__((ext_vector_type(8)))  _Float16 v8h;
typedef __attribute__((ext_vector_type(8)))  float    v8f;
typedef __attribute__((ext_vector_type(4)))  int      v4i;

#define BB 2
#define SS 2048
#define DD 1024
#define HH 16
#define HD 64

#ifndef __has_builtin
#define __has_builtin(x) 0
#endif
#if __has_builtin(__builtin_amdgcn_global_load_async_to_lds_b128)
#define HAVE_ASYNC_LDS 1
#else
#define HAVE_ASYNC_LDS 0
#endif

// ---------------------------------------------------------------------------
// Async global->LDS staging helpers (16 bytes per lane per call)
// ---------------------------------------------------------------------------
__device__ __forceinline__ void stage16(const half_t* g, half_t* l) {
#if HAVE_ASYNC_LDS
  __builtin_amdgcn_global_load_async_to_lds_b128(
      (__attribute__((address_space(1))) v4i*)g,
      (__attribute__((address_space(3))) v4i*)l,
      /*offset=*/0, /*cpol=*/0);
#else
  *(v8h*)l = *(const v8h*)g;          // synchronous fallback
#endif
}

__device__ __forceinline__ void async_wait_all() {
#if HAVE_ASYNC_LDS
#if __has_builtin(__builtin_amdgcn_s_wait_asynccnt)
  __builtin_amdgcn_s_wait_asynccnt(0);
#else
  asm volatile("s_wait_asynccnt 0x0" ::: "memory");
#endif
#endif
}

// ---------------------------------------------------------------------------
// WMMA helpers (ISA 7.12.2 layouts, wave32)
// ---------------------------------------------------------------------------
__device__ __forceinline__ v8f wmma_f16(v16h a, v16h b, v8f c) {
  return __builtin_amdgcn_wmma_f32_16x16x32_f16(
      false, a, false, b, (short)0, c, false, false);
}

// A-matrix 16x32 f16 from row-major [m][k] (stride ld halves).
// lanes 0-15: M=lane, K chunks {0..7,16..23}; lanes 16-31: M=lane-16,
// K chunks {8..15,24..31}.
__device__ __forceinline__ v16h load_frag_a(const half_t* __restrict__ p,
                                            int ld, int lane) {
  const int m  = lane & 15;
  const int hs = (lane >> 4) * 8;
  const half_t* row = p + (size_t)m * ld;
  v8h c0 = *(const v8h*)(row + hs);
  v8h c1 = *(const v8h*)(row + hs + 16);
  v16h a;
#pragma unroll
  for (int i = 0; i < 8; ++i) { a[i] = c0[i]; a[i + 8] = c1[i]; }
  return a;
}

// B-matrix 32x16 f16 from memory row-major over N with K contiguous:
// p[n*ld + k]; lane holds column n=lane&15, K chunks {8h..8h+7, 16+8h..16+8h+7}, h=lane>>4.
__device__ __forceinline__ v16h load_frag_b(const half_t* __restrict__ p,
                                            int ld, int lane) {
  const int n  = lane & 15;
  const int kc = (lane >> 4) * 8;
  const half_t* q = p + (size_t)n * ld + kc;
  v8h c0 = *(const v8h*)(q);
  v8h c1 = *(const v8h*)(q + 16);
  v16h b;
#pragma unroll
  for (int i = 0; i < 8; ++i) { b[i] = c0[i]; b[i + 8] = c1[i]; }
  return b;
}

// C/D 16x16 f32: register r of lane l holds C[r + 8*(l>>4)][l & 15].

// ---------------------------------------------------------------------------
// Kernel 1: f32 -> f16 conversion (activations + concatenated Wq|Wk|Wv)
// ---------------------------------------------------------------------------
__global__ __launch_bounds__(256) void cvt_kernel(
    const float* __restrict__ hs, const float* __restrict__ Wq,
    const float* __restrict__ Wk, const float* __restrict__ Wv,
    half_t* __restrict__ Xh, half_t* __restrict__ Wh) {
  const int idx = blockIdx.x * 256 + threadIdx.x;
  const int nhs = BB * SS * DD;
  if (idx < nhs) {
    Xh[idx] = (half_t)hs[idx];
  } else {
    const int i   = idx - nhs;
    const int sel = i >> 20;
    const int j   = i & ((1 << 20) - 1);
    const float* W = (sel == 0) ? Wq : (sel == 1) ? Wk : Wv;
    Wh[i] = (half_t)W[j];
  }
}

// ---------------------------------------------------------------------------
// Kernel 2: fused QKV GEMM, Y[m,e] = sum_d X[m,d]*W[e,d] + b[e].
// Block = 8 waves; wave w: rows m0..m0+15; block cols nbase..nbase+63.
// Shared 64x32 B tile double-buffered in LDS via async-to-LDS.
// ---------------------------------------------------------------------------
__global__ __launch_bounds__(256) void qkv_gemm_kernel(
    const half_t* __restrict__ Xh, const half_t* __restrict__ Wh,
    const float* __restrict__ bq, const float* __restrict__ bk,
    const float* __restrict__ bv,
    half_t* __restrict__ Qh, half_t* __restrict__ Kh,
    half_t* __restrict__ Vt) {
  __shared__ __align__(16) half_t Bt[2][64 * 32];      // 2 x 4 KB

  const int tid   = threadIdx.x;
  const int lane  = tid & 31;
  const int w     = tid >> 5;
  const int m0    = (blockIdx.x * 8 + w) * 16;
  const int nbase = blockIdx.y * 64;

  // cooperative B-tile copy: thread -> (row n, 8-half chunk) of 64x32 tile
  const int cn = tid >> 2;
  const int ck = (tid & 3) * 8;
  const half_t* wsrc = Wh + (size_t)(nbase + cn) * DD + ck;
  const int     ldst = cn * 32 + ck;

  const half_t* Arow = Xh + (size_t)m0 * DD;
  v8f acc[4] = {v8f{}, v8f{}, v8f{}, v8f{}};

  stage16(wsrc, &Bt[0][ldst]);                         // prefetch k-step 0

  for (int ks = 0; ks < DD / 32; ++ks) {
    const int cur = ks & 1;
    async_wait_all();
    __syncthreads();                                   // Bt[cur] ready
    if (ks + 1 < DD / 32)
      stage16(wsrc + (ks + 1) * 32, &Bt[cur ^ 1][ldst]);

    __builtin_prefetch(Arow + ks * 32 + 128, 0, 0);    // global_prefetch_b8
    v16h a = load_frag_a(Arow + ks * 32, DD, lane);
#pragma unroll
    for (int nf = 0; nf < 4; ++nf) {
      v16h b = load_frag_b(&Bt[cur][nf * 16 * 32], 32, lane);
      acc[nf] = wmma_f16(a, b, acc[nf]);
    }
    __syncthreads();                                   // done reading Bt[cur]
  }

  const int n  = lane & 15;
  const int mo = 8 * (lane >> 4);
#pragma unroll
  for (int nf = 0; nf < 4; ++nf) {
    const int e   = nbase + nf * 16 + n;
    const int sel = e >> 10;                           // 0:q 1:k 2:v
    const int el  = e & 1023;
    const float* bp = (sel == 0) ? bq : (sel == 1) ? bk : bv;
    const float bias = bp[el];
    const int h  = el >> 6;
    const int hd = el & 63;
#pragma unroll
    for (int r = 0; r < 8; ++r) {
      const int m  = m0 + mo + r;
      const int bb = m >> 11;
      const int s  = m & (SS - 1);
      const int bh = bb * HH + h;
      const half_t val = (half_t)(acc[nf][r] + bias);
      if (sel == 0)      Qh[((size_t)bh * SS + s) * HD + hd] = val;
      else if (sel == 1) Kh[((size_t)bh * SS + s) * HD + hd] = val;
      else               Vt[((size_t)bh * HD + hd) * SS + s] = val;
    }
  }
}

// ---------------------------------------------------------------------------
// Kernel 3: RoPE on Q and K (in place, f16), folds 1/sqrt(HD) into Q.
// ---------------------------------------------------------------------------
__global__ __launch_bounds__(256) void rope_kernel(half_t* __restrict__ Qh,
                                                   half_t* __restrict__ Kh) {
  const int idx = blockIdx.x * 256 + threadIdx.x;      // 2^22 threads
  const int i  = idx & 31;
  const int s  = (idx >> 5) & (SS - 1);
  const int bh = (idx >> 16) & 31;
  const int t  = idx >> 21;                            // 0=Q, 1=K
  half_t* buf = (t == 0) ? Qh : Kh;
  const size_t base = ((size_t)bh * SS + s) * HD;
  const float x1 = (float)buf[base + i];
  const float x2 = (float)buf[base + 32 + i];
  const float ang = (float)s * __expf(-(float)i * (9.210340371976184f / 32.0f));
  float sn, cs;
  __sincosf(ang, &sn, &cs);
  const float scale = (t == 0) ? 0.125f : 1.0f;
  buf[base + i]      = (half_t)((x1 * cs - x2 * sn) * scale);
  buf[base + 32 + i] = (half_t)((x1 * sn + x2 * cs) * scale);
}

// ---------------------------------------------------------------------------
// Kernel 4: flash attention.  Block = 8 waves sharing one (b,h); wave w owns
// query rows q0+w*16..+15.  K/V 32-key tiles staged in LDS (double-buffered
// async), consumed by all waves; online softmax; P relayout via per-wave LDS.
// ---------------------------------------------------------------------------
__global__ __launch_bounds__(256) void attn_kernel(
    const half_t* __restrict__ Qh, const half_t* __restrict__ Kh,
    const half_t* __restrict__ Vt, float* __restrict__ out) {
  __shared__ __align__(16) half_t Ktile[2][32 * 64];   // 2 x 4 KB (keys x d)
  __shared__ __align__(16) half_t Vtile[2][64 * 32];   // 2 x 4 KB (d x keys)
  __shared__ __align__(16) half_t plds[8][16][32];     // per-wave P, 8 KB

  const int tid  = threadIdx.x;
  const int lane = tid & 31;
  const int w    = tid >> 5;
  const int bh   = blockIdx.x >> 4;                    // 16 blocks per (b,h)
  const int q0   = (blockIdx.x & 15) * 128 + w * 16;

  const half_t* Kbase = Kh + (size_t)bh * SS * HD;
  const half_t* Vbase = Vt + (size_t)bh * HD * SS;

  // cooperative tile copy indexing (16 B per thread per tensor)
  const half_t* ksrc = Kbase + tid * 8;                // contiguous 4 KB tile
  const int     kdst = tid * 8;
  const int     vrow = tid >> 2;
  const int     vkof = (tid & 3) * 8;
  const half_t* vsrc = Vbase + (size_t)vrow * SS + vkof;
  const int     vdst = vrow * 32 + vkof;

  // Q tile (16x64) as two resident A fragments
  const half_t* Qp = Qh + ((size_t)bh * SS + q0) * HD;
  const v16h aq0 = load_frag_a(Qp, HD, lane);
  const v16h aq1 = load_frag_a(Qp + 32, HD, lane);

  v8f acc0{}, acc1{}, acc2{}, acc3{};
  float rmax[8], rsum[8];
#pragma unroll
  for (int r = 0; r < 8; ++r) { rmax[r] = -1e30f; rsum[r] = 0.0f; }

  half_t* pbuf = &plds[w][0][0];
  const int mo = 8 * (lane >> 4);
  const int nn = lane & 15;

  // prefetch key-tile 0
  stage16(ksrc, &Ktile[0][kdst]);
  stage16(vsrc, &Vtile[0][vdst]);

  for (int kt = 0; kt < SS / 32; ++kt) {
    const int cur = kt & 1;
    const int n0  = kt * 32;
    async_wait_all();
    __syncthreads();                                   // tile kt ready
    if (kt + 1 < SS / 32) {
      stage16(ksrc + (size_t)(n0 + 32) * HD, &Ktile[cur ^ 1][kdst]);
      stage16(vsrc + n0 + 32,                &Vtile[cur ^ 1][vdst]);
    }

    // scores from LDS K tile (ld = 64)
    v8f s0{}, s1{};
    {
      const half_t* Kp = &Ktile[cur][0];
      s0 = wmma_f16(aq0, load_frag_b(Kp, HD, lane), s0);
      s0 = wmma_f16(aq1, load_frag_b(Kp + 32, HD, lane), s0);
      s1 = wmma_f16(aq0, load_frag_b(Kp + 16 * HD, HD, lane), s1);
      s1 = wmma_f16(aq1, load_frag_b(Kp + 16 * HD + 32, HD, lane), s1);
    }

    // online softmax (rows striped across 16-lane halves)
    float p0[8], p1[8];
#pragma unroll
    for (int r = 0; r < 8; ++r) {
      float tm = fmaxf(s0[r], s1[r]);
#pragma unroll
      for (int off = 1; off < 16; off <<= 1)
        tm = fmaxf(tm, __shfl_xor(tm, off, 16));
      const float nm = fmaxf(rmax[r], tm);
      const float sc = __expf(rmax[r] - nm);
      rmax[r] = nm;
      const float e0 = __expf(s0[r] - nm);
      const float e1 = __expf(s1[r] - nm);
      p0[r] = e0; p1[r] = e1;
      float rs = e0 + e1;
#pragma unroll
      for (int off = 1; off < 16; off <<= 1)
        rs += __shfl_xor(rs, off, 16);
      rsum[r] = rsum[r] * sc + rs;
      acc0[r] *= sc; acc1[r] *= sc; acc2[r] *= sc; acc3[r] *= sc;
    }

    // C-layout -> A-layout relayout of P through this wave's LDS region
#pragma unroll
    for (int r = 0; r < 8; ++r) {
      plds[w][mo + r][nn]      = (half_t)p0[r];
      plds[w][mo + r][16 + nn] = (half_t)p1[r];
    }
    const v16h ap = load_frag_a(pbuf, 32, lane);

    // context from LDS V tile (ld = 32)
    const half_t* Vp = &Vtile[cur][0];
    acc0 = wmma_f16(ap, load_frag_b(Vp + 0 * 16 * 32, 32, lane), acc0);
    acc1 = wmma_f16(ap, load_frag_b(Vp + 1 * 16 * 32, 32, lane), acc1);
    acc2 = wmma_f16(ap, load_frag_b(Vp + 2 * 16 * 32, 32, lane), acc2);
    acc3 = wmma_f16(ap, load_frag_b(Vp + 3 * 16 * 32, 32, lane), acc3);

    __syncthreads();                                   // done reading tile kt
  }

  // epilogue: divide by row sums, write (b, s, h*64 + hd) f32
  const int b = bh >> 4;
  const int h = bh & 15;
#pragma unroll
  for (int r = 0; r < 8; ++r) {
    const int s = q0 + mo + r;
    const float inv = 1.0f / rsum[r];
    const size_t base = ((size_t)b * SS + s) * DD + h * HD + nn;
    out[base]      = acc0[r] * inv;
    out[base + 16] = acc1[r] * inv;
    out[base + 32] = acc2[r] * inv;
    out[base + 48] = acc3[r] * inv;
  }
}

// ---------------------------------------------------------------------------
// Launch
// ---------------------------------------------------------------------------
extern "C" void kernel_launch(void* const* d_in, const int* in_sizes, int n_in,
                              void* d_out, int out_size, void* d_ws,
                              size_t ws_size, hipStream_t stream) {
  const float* hs = (const float*)d_in[0];
  const float* Wq = (const float*)d_in[1];
  const float* bq = (const float*)d_in[2];
  const float* Wk = (const float*)d_in[3];
  const float* bk = (const float*)d_in[4];
  const float* Wv = (const float*)d_in[5];
  const float* bv = (const float*)d_in[6];
  float* out = (float*)d_out;

  half_t* Xh = (half_t*)d_ws;
  half_t* Wh = Xh + (size_t)BB * SS * DD;
  half_t* Qh = Wh + (size_t)3 * DD * DD;
  half_t* Kh = Qh + (size_t)BB * HH * SS * HD;
  half_t* Vt = Kh + (size_t)BB * HH * SS * HD;

  const int cvt_total = BB * SS * DD + 3 * DD * DD;
  cvt_kernel<<<cvt_total / 256, 256, 0, stream>>>(hs, Wq, Wk, Wv, Xh, Wh);

  qkv_gemm_kernel<<<dim3(32, 48), 256, 0, stream>>>(Xh, Wh, bq, bk, bv,
                                                    Qh, Kh, Vt);

  rope_kernel<<<(1 << 22) / 256, 256, 0, stream>>>(Qh, Kh);

  attn_kernel<<<512, 256, 0, stream>>>(Qh, Kh, Vt, out);
}
